// LSTM_69767448756334
// MI455X (gfx1250) — compile-verified, matrix operand order checked by reading, not run on hardware
//

#include <hip/hip_runtime.h>
#include <hip/hip_bf16.h>

// ---------------------------------------------------------------------------
// LSTM forward on MI455X (gfx1250), wave32, bf16 WMMA (f32 accum), LDS-staged
// weights, scheduler-directed software-pipelined inner GEMM.
//
//   gates(32 x 2048) = [x_t | h_{t-1}] (32 x 1024) @ WT^T (+ b)
//
// WT = bf16 combined pre-transposed weight [Wx;Wh]^T, shape (2048 x 1024).
// 32 persistent workgroups (1 per h_tile of 16 columns), 2 waves each
// (m_tile 0/1 of the N=32 batch rows). Each WG copies its 128 KB WT slice
// into LDS once (rows padded 1024->1032 bf16 for bank-conflict-free b128
// reads). Per step: one uniform 32-chunk K loop, fully unrolled, pipelined
// one chunk ahead via sched_group_barrier (DS/VMEM loads of chunk k+1 before
// WMMAs of chunk k). x_{t+1}/h_{t-1} rows prefetched at WGP scope via inline
// asm. c-state lives in registers; h ping-pongs through a global bf16 buffer
// with one device-wide atomic barrier per timestep.
// ---------------------------------------------------------------------------

typedef __attribute__((ext_vector_type(16))) __bf16 v16bf;
typedef __attribute__((ext_vector_type(8)))  __bf16 v8bf;
typedef __attribute__((ext_vector_type(8)))  float  v8f;

#define N_  32
#define T_  1024
#define D_  512
#define H_  512
#define KTOT 1024            // D_ + H_
#define NWG  32              // persistent blocks (one per h_tile)
#define ROWP 1032            // padded LDS row stride in bf16 elements
#define LDS_ELEMS (4 * 16 * ROWP)          // 66048 bf16
#define LDS_BYTES (LDS_ELEMS * 2)          // 132096 B (dynamic shared)

#if __has_builtin(__builtin_amdgcn_sched_group_barrier)
#define SGB(mask, cnt) __builtin_amdgcn_sched_group_barrier((mask), (cnt), 0)
#else
#define SGB(mask, cnt)
#endif
// sched_group_barrier masks: 0x8 = MFMA/WMMA, 0x20 = VMEM read, 0x100 = DS read

// WGP-scope prefetch (fills all cache levels; builtin only reaches SE scope).
#define PREFETCH_WGP(PTR, BOFF)                                               \
    asm volatile("global_prefetch_b8 %0, off offset:" #BOFF                   \
                 :: "v"(PTR) : "memory")

// One 16x32 bf16 A fragment (wave32 layout, CDNA5 ISA 7.12.2):
//   lanes 0-15 : row = lane,    K chunks [kb+0..7]  and [kb+16..23]
//   lanes16-31 : row = lane-16, K chunks [kb+8..15] and [kb+24..31]
__device__ __forceinline__ v16bf load_frag_g(const __bf16* __restrict__ rowptr,
                                             int kb, int hioff) {
    union { v16bf v; v8bf p[2]; } u;
    u.p[0] = *(const v8bf*)(rowptr + kb + hioff);        // global_load_b128
    u.p[1] = *(const v8bf*)(rowptr + kb + 16 + hioff);   // global_load_b128
    return u.v;
}

// 4 B-fragments (one per gate) for one K-chunk from LDS.
struct B4 { v16bf b0, b1, b2, b3; };
__device__ __forceinline__ B4 load_b4(const __bf16* lp, const __bf16* hp,
                                      int koff) {
    union { v16bf v; v8bf p[2]; } u0, u1, u2, u3;
    u0.p[0] = *(const v8bf*)(lp + koff);
    u0.p[1] = *(const v8bf*)(lp + koff + 16);
    u1.p[0] = *(const v8bf*)(lp + 16 * ROWP + koff);
    u1.p[1] = *(const v8bf*)(lp + 16 * ROWP + koff + 16);
    u2.p[0] = *(const v8bf*)(hp + koff);
    u2.p[1] = *(const v8bf*)(hp + koff + 16);
    u3.p[0] = *(const v8bf*)(hp + 16 * ROWP + koff);
    u3.p[1] = *(const v8bf*)(hp + 16 * ROWP + koff + 16);
    B4 r; r.b0 = u0.v; r.b1 = u1.v; r.b2 = u2.v; r.b3 = u3.v;
    return r;
}

__device__ __forceinline__ float sigmoidf_(float x) {
    return 1.0f / (1.0f + __expf(-x));
}

// Branch-free tanh via exp; no overflow (argument of exp is <= 0).
__device__ __forceinline__ float fast_tanh(float x) {
    float a = __builtin_fabsf(x);
    float e = __expf(-2.0f * a);
    float th = (1.0f - e) / (1.0f + e);
    return __builtin_copysignf(th, x);
}

// ---------------------------- prep kernels ---------------------------------

__global__ void k_cvt_x_bf16(const float* __restrict__ x,
                             __bf16* __restrict__ xb, int n) {
    int i = blockIdx.x * blockDim.x + threadIdx.x;
    int stride = gridDim.x * blockDim.x;
    for (; i < n; i += stride) xb[i] = (__bf16)x[i];
}

// WT[n][k] (n in [0,2048), k in [0,1024)) = k<512 ? Wx[k][n] : Wh[k-512][n]
__global__ void k_build_wt(const float* __restrict__ Wx,
                           const float* __restrict__ Wh,
                           __bf16* __restrict__ wt) {
    int i = blockIdx.x * blockDim.x + threadIdx.x;
    int stride = gridDim.x * blockDim.x;
    const int total = 2048 * KTOT;
    for (; i < total; i += stride) {
        int n = i >> 10;           // / 1024
        int k = i & (KTOT - 1);    // % 1024
        float v = (k < D_) ? Wx[(size_t)k * 2048 + n]
                           : Wh[(size_t)(k - D_) * 2048 + n];
        wt[(size_t)n * KTOT + k] = (__bf16)v;
    }
}

__global__ void k_init_h(const float* __restrict__ h0,
                         __bf16* __restrict__ hbuf0,
                         unsigned* __restrict__ counter) {
    int i = blockIdx.x * blockDim.x + threadIdx.x;
    int stride = gridDim.x * blockDim.x;
    for (; i < N_ * H_; i += stride) hbuf0[i] = (__bf16)h0[i];
    if (blockIdx.x == 0 && threadIdx.x == 0) *counter = 0u;   // reset barrier
}

// -------------------------- persistent kernel ------------------------------

__global__ __launch_bounds__(64, 1)
void k_lstm_persistent(const __bf16* __restrict__ xb,     // (N,T,D) bf16
                       const __bf16* __restrict__ wt,     // (2048,1024) bf16
                       const float*  __restrict__ bias,   // (2048,) f32
                       __bf16* __restrict__ hbuf,         // 2 x (N,H) bf16
                       unsigned* __restrict__ counter,
                       float* __restrict__ out) {         // (N,T,H) f32
    extern __shared__ __bf16 lds[];                       // [4][16][ROWP]

    const int lane   = threadIdx.x & 31;
    const int m_tile = threadIdx.x >> 5;                  // wave = m_tile (0/1)
    const int h_tile = blockIdx.x;                        // 0..31
    const int col    = lane & 15;
    const int hioff  = (lane >> 4) << 3;                  // 0 or 8

    // ---- stage this WG's WT slice into LDS (once) ----
    // slice rows: gate g, row r -> global row g*512 + h_tile*16 + r
    // LDS layout: [(g*16 + r) * ROWP + k], k in [0,1024)
    {
        const int tid = threadIdx.x;                      // 0..63
        for (int c = tid; c < 4 * 16 * 128; c += 64) {    // 16B chunks
            int g  = c >> 11;                             // /2048
            int rm = c & 2047;
            int r  = rm >> 7;                             // /128
            int ck = rm & 127;
            const v8bf* src = (const v8bf*)(wt +
                (size_t)(g * H_ + h_tile * 16 + r) * KTOT + ck * 8);
            *(v8bf*)&lds[(g * 16 + r) * ROWP + ck * 8] = *src;
        }
    }
    __syncthreads();

    // Two LDS pointer bases; gates 1/3 reachable via +16*ROWP immediate.
    const __bf16* lp = lds + (col * ROWP + hioff);        // gate 0 base
    const __bf16* hp = lp + 32 * ROWP;                    // gate 2 base

    // A row: batch row n = m_tile*16 + col.
    const int a_row = m_tile * 16 + col;
    const __bf16* x_row = xb + (size_t)a_row * T_ * D_;
    const __bf16* h_row_base = hbuf + (size_t)a_row * H_;

    float bg[4];
#pragma unroll
    for (int g = 0; g < 4; ++g) bg[g] = bias[g * H_ + h_tile * 16 + col];

    const int h_col = h_tile * 16 + col;
    v8f c = {0.f, 0.f, 0.f, 0.f, 0.f, 0.f, 0.f, 0.f};

    for (int t = 0; t < T_; ++t) {
        const __bf16* x_t  = x_row + (size_t)t * D_;
        const __bf16* h_rd = h_row_base + (size_t)(t & 1) * (N_ * H_);
        __bf16* h_wr = hbuf + (size_t)((t + 1) & 1) * (N_ * H_);

        // Prefetch h_{t-1} rows right after the barrier at WGP scope (fills
        // all cache levels); the x-half of the GEMM hides the round trip.
        // Lanes 0-15 cover bytes [0,512) of their row, lanes 16-31 [512,1024).
        {
            const __bf16* ph = h_rd + (hioff ? 256 : 0);
            PREFETCH_WGP(ph, 0);
            PREFETCH_WGP(ph, 128);
            PREFETCH_WGP(ph, 256);
            PREFETCH_WGP(ph, 384);
        }
        // Prefetch next timestep's x rows (independent of recurrence).
        if (t + 1 < T_) {
            const __bf16* nx = x_row + (size_t)(t + 1) * D_ + (hioff ? 256 : 0);
            PREFETCH_WGP(nx, 0);
            PREFETCH_WGP(nx, 128);
            PREFETCH_WGP(nx, 256);
            PREFETCH_WGP(nx, 384);
        }

        v8f acc[4];
#pragma unroll
        for (int g = 0; g < 4; ++g)
            acc[g] = (v8f){0.f, 0.f, 0.f, 0.f, 0.f, 0.f, 0.f, 0.f};

        // ---- GEMM over K = [x_t | h_{t-1}], 32 chunks of 32, fully
        // unrolled. LDS K offset is kc*32 throughout; the A base flips from
        // x_t to h_rd at kc==16 (compile-time under unroll).
        v16bf a_cur = load_frag_g(x_t, 0, hioff);
        B4    b_cur = load_b4(lp, hp, 0);
#pragma unroll
        for (int kc = 0; kc < 32; ++kc) {
            v16bf a_nxt; B4 b_nxt;
            if (kc < 31) {
                const int kn = kc + 1;
                const __bf16* ab = (kn < 16) ? x_t : (h_rd - D_);
                a_nxt = load_frag_g(ab, kn * 32, hioff);
                b_nxt = load_b4(lp, hp, kn * 32);
            }
            acc[0] = __builtin_amdgcn_wmma_f32_16x16x32_bf16(
                false, a_cur, false, b_cur.b0, (short)0, acc[0], false, false);
            acc[1] = __builtin_amdgcn_wmma_f32_16x16x32_bf16(
                false, a_cur, false, b_cur.b1, (short)0, acc[1], false, false);
            acc[2] = __builtin_amdgcn_wmma_f32_16x16x32_bf16(
                false, a_cur, false, b_cur.b2, (short)0, acc[2], false, false);
            acc[3] = __builtin_amdgcn_wmma_f32_16x16x32_bf16(
                false, a_cur, false, b_cur.b3, (short)0, acc[3], false, false);
            if (kc < 31) { a_cur = a_nxt; b_cur = b_nxt; }
        }

        // Scheduler pipeline directive for the whole unrolled GEMM region:
        // loads of chunk k+1 (8 DS reads + 2 VMEM reads) are grouped BEFORE
        // the 4 WMMAs of chunk k, giving every load a 4-WMMA shadow and the
        // B registers enough WAR distance to avoid hazard NOPs.
        SGB(0x100, 8);  SGB(0x020, 2);          // chunk 0 loads
#pragma unroll
        for (int kc = 0; kc < 31; ++kc) {
            SGB(0x100, 8);                      // chunk kc+1 DS reads
            SGB(0x020, 2);                      // chunk kc+1 VMEM reads
            SGB(0x008, 4);                      // chunk kc WMMAs
        }
        SGB(0x008, 4);                          // chunk 31 WMMAs

        // Cell update. C/D layout: VGPR r -> lanes 0-15 M=r, lanes 16-31 M=8+r.
#pragma unroll
        for (int r = 0; r < 8; ++r) {
            float i_ = sigmoidf_(acc[0][r] + bg[0]);
            float f_ = sigmoidf_(acc[1][r] + bg[1]);
            float o_ = sigmoidf_(acc[2][r] + bg[2]);
            float g_ = fast_tanh(acc[3][r] + bg[3]);
            float cn = f_ * c[r] + i_ * g_;
            c[r] = cn;
            float hn = o_ * fast_tanh(cn);
            int n_row = m_tile * 16 + (hioff ? 8 : 0) + r;
            out[((size_t)n_row * T_ + t) * H_ + h_col] = hn;
            h_wr[(size_t)n_row * H_ + h_col] = (__bf16)hn;
        }

        // ---- device-wide barrier (monotonic counter) ----
        __threadfence();
        __syncthreads();
        if (threadIdx.x == 0) {
            __hip_atomic_fetch_add(counter, 1u, __ATOMIC_ACQ_REL,
                                   __HIP_MEMORY_SCOPE_AGENT);
            const unsigned target = (unsigned)(t + 1) * NWG;
            while (__hip_atomic_load(counter, __ATOMIC_ACQUIRE,
                                     __HIP_MEMORY_SCOPE_AGENT) < target) {
                __builtin_amdgcn_s_sleep(2);
            }
        }
        __syncthreads();
    }
}

// ------------------------------- launcher ----------------------------------

extern "C" void kernel_launch(void* const* d_in, const int* in_sizes, int n_in,
                              void* d_out, int out_size, void* d_ws, size_t ws_size,
                              hipStream_t stream) {
    const float* x  = (const float*)d_in[0];   // (N,T,D)
    const float* h0 = (const float*)d_in[1];   // (N,H)
    const float* Wx = (const float*)d_in[2];   // (D,4H)
    const float* Wh = (const float*)d_in[3];   // (H,4H)
    const float* b  = (const float*)d_in[4];   // (4H,)
    float* out = (float*)d_out;                // (N,T,H)

    // Workspace layout (bytes):
    //   xb   : N*T*D bf16     = 33,554,432
    //   wt   : 2048*1024 bf16 =  4,194,304
    //   hbuf : 2*N*H bf16     =     65,536
    //   cnt  : 4
    char* ws = (char*)d_ws;
    __bf16* xb    = (__bf16*)ws;
    __bf16* wt    = (__bf16*)(ws + (size_t)33554432);
    __bf16* hbuf  = (__bf16*)(ws + (size_t)33554432 + 4194304);
    unsigned* cnt = (unsigned*)(ws + (size_t)33554432 + 4194304 + 65536);

    const int nx = N_ * T_ * D_;
    k_cvt_x_bf16<<<4096, 256, 0, stream>>>(x, xb, nx);
    k_build_wt  <<<2048, 256, 0, stream>>>(Wx, Wh, wt);
    k_init_h    <<<64,   256, 0, stream>>>(h0, hbuf, cnt);

    k_lstm_persistent<<<dim3(NWG), dim3(64), LDS_BYTES, stream>>>(
        xb, wt, b, hbuf, cnt, out);
}
